// FiLM_FNO_47682726920687
// MI455X (gfx1250) — compile-verified, matrix operand order checked by reading, not run on hardware
//
#include <hip/hip_runtime.h>
#include <hip/hip_bf16.h>
#include <math.h>

typedef __bf16 bf16_t;
typedef __attribute__((ext_vector_type(16))) bf16_t v16bf;
typedef __attribute__((ext_vector_type(8)))  bf16_t v8bf;
typedef __attribute__((ext_vector_type(8)))  float  v8f;

#define NSd 64
#define NGd 3
#define Wd  128
#define Ld  1024
#define Md  32
#define TDd 64
#define CDd 128
#define NBd 4
#define Bd  256

union V16U { v16bf v; v8bf h[2]; };

__device__ __forceinline__ float gelu_f(float x) {
  return 0.5f * x * (1.0f + erff(x * 0.70710678f));
}

// A/B fragment load from bf16 matrix stored [row][k] (row = M for A, N for B),
// contiguous in k, leading dim ld. Lane layout per CDNA5 ISA 16-bit A matrix:
// lanes 0-15: K runs {0..7},{16..23}; lanes 16-31: K runs {8..15},{24..31}.
__device__ __forceinline__ v16bf ldfrag_bf16(const bf16_t* base, int ld) {
  const int lane = threadIdx.x & 31;
  const int r  = lane & 15;
  const int k0 = (lane & 16) ? 8 : 0;
  const bf16_t* p = base + r * ld + k0;
  V16U t;
  t.h[0] = *reinterpret_cast<const v8bf*>(p);
  t.h[1] = *reinterpret_cast<const v8bf*>(p + 16);
  return t.v;
}

__device__ __forceinline__ v8f wmma_bf16(v16bf a, v16bf b, v8f c) {
  return __builtin_amdgcn_wmma_f32_16x16x32_bf16(false, a, false, b, (short)0, c,
                                                 false, false);
}

// ---------------- prep: DFT / iDFT bases + positional encoding ----------------
__global__ void k_basis(bf16_t* __restrict__ dftB, bf16_t* __restrict__ idtB,
                        float* __restrict__ pe) {
  int l = blockIdx.x * blockDim.x + threadIdx.x;
  if (l >= Ld) return;
  const float w0 = 6.28318530717958647692f / (float)Ld;
  for (int j = 0; j < 64; ++j) {
    int x = j & 31;
    float th = w0 * (float)(x * l);
    float fwd = (j < 32) ? cosf(th) : -sinf(th);
    dftB[j * Ld + l] = (bf16_t)fwd;            // [n=j][k=l]
    float sc = (x == 0 ? 1.0f : 2.0f) * (1.0f / (float)Ld);
    float inv = (j < 32) ? sc * cosf(th) : -sc * sinf(th);
    idtB[l * 64 + j] = (bf16_t)inv;            // [n=l][k=j]
  }
  const float c = -logf(10000.0f) / (float)Wd;
  for (int w = 0; w < Wd; ++w) {
    float div = expf((float)(2 * (w >> 1)) * c);
    float a = (float)l * div;
    pe[w * Ld + l] = (w & 1) ? cosf(a) : sinf(a);
  }
}

// ---------------- prep: weight conversion / transposition to bf16 ------------
__global__ void k_wprep(const float* __restrict__ spec_w,
                        const float* __restrict__ conv_w,
                        const float* __restrict__ proj_w1,
                        const float* __restrict__ proj_w2,
                        bf16_t* __restrict__ Wre, bf16_t* __restrict__ Wim,
                        bf16_t* __restrict__ Wimn, bf16_t* __restrict__ cwbf,
                        bf16_t* __restrict__ p1bf, bf16_t* __restrict__ p2bf) {
  const long NSPEC = (long)NBd * Md * Wd * Wd;     // [blk][x][o][i]
  const long NCW   = (long)NBd * Wd * Wd;          // [blk][o][c]
  const long NP1   = 64 * 128;
  const long NP2   = 64 * 64;
  long idx = (long)blockIdx.x * blockDim.x + threadIdx.x;
  if (idx < NSPEC) {
    long i = idx & 127, o = (idx >> 7) & 127, x = (idx >> 14) & 31, blk = idx >> 19;
    long src = (((blk * Wd + i) * Wd + o) * Md + x) * 2;  // spec_w (NB,i,o,M,2)
    float re = spec_w[src], im = spec_w[src + 1];
    Wre[idx]  = (bf16_t)re;
    Wim[idx]  = (bf16_t)im;
    Wimn[idx] = (bf16_t)(-im);
  } else if (idx < NSPEC + NCW) {
    long e = idx - NSPEC;
    cwbf[e] = (bf16_t)conv_w[e];
  } else if (idx < NSPEC + NCW + NP1) {
    long e = idx - NSPEC - NCW;
    p1bf[e] = (bf16_t)proj_w1[e];
  } else if (idx < NSPEC + NCW + NP1 + NP2) {
    long e = idx - NSPEC - NCW - NP1;
    p2bf[e] = (bf16_t)proj_w2[e];
  }
}

// ---------------- conditioning MLP + lift (tiny) ------------------------------
__global__ void k_cond(const float* __restrict__ x, const float* __restrict__ tss,
                       const float* __restrict__ w1, const float* __restrict__ b1,
                       const float* __restrict__ w2, const float* __restrict__ b2,
                       const float* __restrict__ lw, const float* __restrict__ lb,
                       float* __restrict__ lifted) {
  __shared__ float cin[NGd + TDd];
  __shared__ float hbuf[CDd];
  __shared__ float vbuf[NSd + CDd];
  const int b = blockIdx.x;
  const int tid = threadIdx.x;  // 128
  const float* xr = x + b * (NSd + NGd + 1);
  if (tid < NGd) cin[tid] = xr[NSd + tid];
  float t = xr[NSd + NGd] * tss[0];
  if (tid < TDd) {
    int j = tid & 31;
    float invf = expf(-logf(10000.0f) * (float)j * (1.0f / 31.0f));
    float a = t * invf;
    cin[NGd + tid] = (tid < 32) ? sinf(a) : cosf(a);
  }
  if (tid < NSd) vbuf[tid] = xr[tid];
  __syncthreads();
  float acc = b1[tid];
  for (int k = 0; k < NGd + TDd; ++k) acc += w1[tid * (NGd + TDd) + k] * cin[k];
  hbuf[tid] = gelu_f(acc);
  __syncthreads();
  float acc2 = b2[tid];
  for (int k = 0; k < CDd; ++k) acc2 += w2[tid * CDd + k] * hbuf[k];
  vbuf[NSd + tid] = acc2;
  __syncthreads();
  float acc3 = lb[tid];
  for (int k = 0; k < NSd + CDd; ++k) acc3 += lw[tid * (NSd + CDd) + k] * vbuf[k];
  lifted[b * Wd + tid] = gelu_f(acc3);
}

// ------- fr0 = lifted + POS, cast to bf16, dual layout [b][c][l] & [b][l][c] --
__global__ void k_fr0(const float* __restrict__ lifted, const float* __restrict__ pe,
                      bf16_t* __restrict__ fr_il, bf16_t* __restrict__ fr_li) {
  long idx = (long)blockIdx.x * blockDim.x + threadIdx.x;
  int  l = (int)(idx & (Ld - 1));
  long row = idx >> 10;             // b*128 + w
  int  w = (int)(row & (Wd - 1));
  long b = row >> 7;
  float v = lifted[row] + pe[w * Ld + l];
  fr_il[idx] = (bf16_t)v;
  fr_li[(b * Ld + l) * Wd + w] = (bf16_t)v;
}

// -------- forward truncated DFT: FT[j][b*i] = fr x dftB (pipelined) -----------
// One wave handles TWO 16-row tiles (A reuse of B fragments), prefetches the
// next k-step while 8 WMMAs execute. launch_bounds widens the VGPR budget so
// both pipeline stages stay in registers (no scratch spills).
__global__ void __launch_bounds__(256, 1)
k_dft(const bf16_t* __restrict__ fr, const bf16_t* __restrict__ dftB,
      bf16_t* __restrict__ FT) {
  const int wid = threadIdx.x >> 5;
  const int lane = threadIdx.x & 31;
  const int rt = blockIdx.x * 8 + wid;  // 0..1023, each covers 32 rows of (b,i)
  const bf16_t* a0base = fr + (size_t)rt * 32 * Ld;
  const bf16_t* a1base = a0base + 16 * Ld;
  v8f acc[8] = {};
  v16bf a0 = ldfrag_bf16(a0base, Ld);
  v16bf a1 = ldfrag_bf16(a1base, Ld);
  v16bf b0 = ldfrag_bf16(dftB + 0 * 16 * Ld, Ld);
  v16bf b1 = ldfrag_bf16(dftB + 1 * 16 * Ld, Ld);
  v16bf b2 = ldfrag_bf16(dftB + 2 * 16 * Ld, Ld);
  v16bf b3 = ldfrag_bf16(dftB + 3 * 16 * Ld, Ld);
  for (int kt = 0; kt < 32; ++kt) {
    v16bf na0 = a0, na1 = a1, nb0 = b0, nb1 = b1, nb2 = b2, nb3 = b3;
    if (kt < 31) {
      int ko = (kt + 1) * 32;
      na0 = ldfrag_bf16(a0base + ko, Ld);
      na1 = ldfrag_bf16(a1base + ko, Ld);
      nb0 = ldfrag_bf16(dftB + 0 * 16 * Ld + ko, Ld);
      nb1 = ldfrag_bf16(dftB + 1 * 16 * Ld + ko, Ld);
      nb2 = ldfrag_bf16(dftB + 2 * 16 * Ld + ko, Ld);
      nb3 = ldfrag_bf16(dftB + 3 * 16 * Ld + ko, Ld);
    }
    acc[0] = wmma_bf16(a0, b0, acc[0]);
    acc[1] = wmma_bf16(a0, b1, acc[1]);
    acc[2] = wmma_bf16(a0, b2, acc[2]);
    acc[3] = wmma_bf16(a0, b3, acc[3]);
    acc[4] = wmma_bf16(a1, b0, acc[4]);
    acc[5] = wmma_bf16(a1, b1, acc[5]);
    acc[6] = wmma_bf16(a1, b2, acc[6]);
    acc[7] = wmma_bf16(a1, b3, acc[7]);
    a0 = na0; a1 = na1; b0 = nb0; b1 = nb1; b2 = nb2; b3 = nb3;
  }
  const int n = lane & 15, m0 = (lane & 16) ? 8 : 0;
#pragma unroll
  for (int mt = 0; mt < 2; ++mt) {
#pragma unroll
    for (int nt = 0; nt < 4; ++nt) {
      bf16_t* p = FT + (size_t)(nt * 16 + n) * (Bd * Wd) + rt * 32 + mt * 16 + m0;
      v8f& a = acc[mt * 4 + nt];
      v8bf pk;
#pragma unroll
      for (int r2 = 0; r2 < 8; ++r2) pk[r2] = (bf16_t)a[r2];
      *reinterpret_cast<v8bf*>(p) = pk;   // 8 consecutive rows, one b128 store
    }
  }
}

// ---------------- per-mode complex channel mixing -----------------------------
__global__ void __launch_bounds__(256, 1)
k_mix(const bf16_t* __restrict__ FT, const bf16_t* __restrict__ Wre,
      const bf16_t* __restrict__ Wim, const bf16_t* __restrict__ Wimn,
      bf16_t* __restrict__ OUT) {
  const int wid = threadIdx.x >> 5;
  const int lane = threadIdx.x & 31;
  int t = blockIdx.x * 8 + wid;  // 0..4095
  int x = t & 31;
  int ot = (t >> 5) & 7;
  int bt = t >> 8;
  v8f accR = {}, accI = {};
  const bf16_t* ArB = FT + (size_t)x * (Bd * Wd) + (size_t)bt * 16 * Wd;
  const bf16_t* AiB = FT + (size_t)(32 + x) * (Bd * Wd) + (size_t)bt * 16 * Wd;
  const size_t woff = (size_t)x * Wd * Wd + (size_t)ot * 16 * Wd;
#pragma unroll
  for (int kt = 0; kt < 4; ++kt) {
    v16bf ar = ldfrag_bf16(ArB + kt * 32, Wd);
    v16bf ai = ldfrag_bf16(AiB + kt * 32, Wd);
    v16bf bre  = ldfrag_bf16(Wre  + woff + kt * 32, Wd);
    v16bf bim  = ldfrag_bf16(Wim  + woff + kt * 32, Wd);
    v16bf bimn = ldfrag_bf16(Wimn + woff + kt * 32, Wd);
    accR = wmma_bf16(ar, bre,  accR);   // Re*Re
    accR = wmma_bf16(ai, bimn, accR);   // -Im*Im
    accI = wmma_bf16(ar, bim,  accI);   // Re*Im
    accI = wmma_bf16(ai, bre,  accI);   // Im*Re
  }
  const int n = lane & 15, m0 = (lane & 16) ? 8 : 0;
#pragma unroll
  for (int r2 = 0; r2 < 8; ++r2) {
    size_t row = (size_t)(bt * 16 + m0 + r2) * Wd + (ot * 16 + n);
    OUT[row * 64 + x]      = (bf16_t)accR[r2];
    OUT[row * 64 + 32 + x] = (bf16_t)accI[r2];
  }
}

// ------- fused iDFT + 1x1 conv + bias + channel-LayerNorm + GELU --------------
// Conv B operand comes straight from the [b][l][c] copy (no LDS transpose).
// Updates both fr layouts in place (reads/writes of this workgroup's own tile
// are separated by the LayerNorm barrier).
__global__ void k_block(const bf16_t* __restrict__ OUT,
                        const bf16_t* __restrict__ idtB,
                        const bf16_t* __restrict__ cwbf,
                        const float* __restrict__ cb, const float* __restrict__ g,
                        const float* __restrict__ bt_,
                        bf16_t* __restrict__ fr_il, bf16_t* __restrict__ fr_li) {
  __shared__ float red_s[16], red_s2[16];
  const int b  = blockIdx.x >> 6;
  const int lt = blockIdx.x & 63;
  const int tid = threadIdx.x;  // 256 = 8 waves, one 16-channel tile each
  const int wid = tid >> 5, lane = tid & 31;
  if (tid < 16) { red_s[tid] = 0.f; red_s2[tid] = 0.f; }
  const int ot = wid;
  v8f acc = {};
  // x1 = OUT x iDFT basis  (K = 64)
  const bf16_t* Ab = OUT + (size_t)(b * Wd + ot * 16) * 64;
#pragma unroll
  for (int kt = 0; kt < 2; ++kt) {
    v16bf a  = ldfrag_bf16(Ab + kt * 32, 64);
    v16bf bb = ldfrag_bf16(idtB + lt * 16 * 64 + kt * 32, 64);
    acc = wmma_bf16(a, bb, acc);
  }
  // x2 = conv_w x fr       (K = 128), B direct from fr_li
  const bf16_t* Bb = fr_li + ((size_t)b * Ld + lt * 16) * Wd;
#pragma unroll
  for (int kt = 0; kt < 4; ++kt) {
    v16bf a  = ldfrag_bf16(cwbf + ot * 16 * Wd + kt * 32, Wd);
    v16bf bb = ldfrag_bf16(Bb + kt * 32, Wd);
    acc = wmma_bf16(a, bb, acc);
  }
  const int n = lane & 15, m0 = (lane & 16) ? 8 : 0;
#pragma unroll
  for (int r2 = 0; r2 < 8; ++r2) acc[r2] += cb[ot * 16 + m0 + r2];
  // LayerNorm over 128-channel axis
  float s = 0.f, s2 = 0.f;
#pragma unroll
  for (int r2 = 0; r2 < 8; ++r2) { float v = acc[r2]; s += v; s2 += v * v; }
  s  += __shfl_xor(s, 16, 32);
  s2 += __shfl_xor(s2, 16, 32);
  __syncthreads();  // zeroing (and all conv reads) complete before adds/stores
  if (lane < 16) { atomicAdd(&red_s[n], s); atomicAdd(&red_s2[n], s2); }
  __syncthreads();
  float mu   = red_s[n] * (1.0f / Wd);
  float var  = red_s2[n] * (1.0f / Wd) - mu * mu;
  float rstd = rsqrtf(var + 1e-5f);
  float vals[8];
  v8bf pk;
#pragma unroll
  for (int r2 = 0; r2 < 8; ++r2) {
    int o = ot * 16 + m0 + r2;
    float y = (acc[r2] - mu) * rstd * g[o] + bt_[o];
    vals[r2] = gelu_f(y);
    pk[r2] = (bf16_t)vals[r2];
  }
  // packed contiguous store into [b][l][c] layout (8 consecutive channels)
  *reinterpret_cast<v8bf*>(fr_li + ((size_t)b * Ld + lt * 16 + n) * Wd +
                           ot * 16 + m0) = pk;
#pragma unroll
  for (int r2 = 0; r2 < 8; ++r2) {
    int o = ot * 16 + m0 + r2;
    fr_il[((size_t)(b * Wd + o) << 10) + lt * 16 + n] = (bf16_t)vals[r2];
  }
}

// ---------------- final projections + mean over L -----------------------------
__global__ void k_proj(const bf16_t* __restrict__ fr_li,
                       const bf16_t* __restrict__ p1bf,
                       const float* __restrict__ pb1,
                       const bf16_t* __restrict__ p2bf,
                       const float* __restrict__ pb2,
                       const float* __restrict__ x, float* __restrict__ out) {
  __shared__ bf16_t h2t[16 * 64];    // [l][o1]
  const int b = blockIdx.x;
  const int tid = threadIdx.x;  // 128 = 4 waves
  const int wid = tid >> 5, lane = tid & 31;
  const int ot = wid;
  const int n = lane & 15, m0 = (lane & 16) ? 8 : 0;
  v8f accD = {};
  for (int lt = 0; lt < 64; ++lt) {
    const bf16_t* Bb = fr_li + ((size_t)b * Ld + lt * 16) * Wd;
    v8f a1 = {};
#pragma unroll
    for (int kt = 0; kt < 4; ++kt) {
      v16bf a  = ldfrag_bf16(p1bf + ot * 16 * Wd + kt * 32, Wd);
      v16bf bb = ldfrag_bf16(Bb + kt * 32, Wd);
      a1 = wmma_bf16(a, bb, a1);
    }
#pragma unroll
    for (int r2 = 0; r2 < 8; ++r2) {
      int o1 = ot * 16 + m0 + r2;
      h2t[n * 64 + o1] = (bf16_t)gelu_f(a1[r2] + pb1[o1]);
    }
    __syncthreads();
#pragma unroll
    for (int kt = 0; kt < 2; ++kt) {
      v16bf a  = ldfrag_bf16(p2bf + ot * 16 * 64 + kt * 32, 64);
      v16bf bb = ldfrag_bf16(&h2t[0] + kt * 32, 64);
      accD = wmma_bf16(a, bb, accD);
    }
    __syncthreads();
  }
#pragma unroll
  for (int r2 = 0; r2 < 8; ++r2) {
    float v = accD[r2];
    v += __shfl_xor(v, 1, 32);
    v += __shfl_xor(v, 2, 32);
    v += __shfl_xor(v, 4, 32);
    v += __shfl_xor(v, 8, 32);
    if (n == 0) {
      int o = ot * 16 + m0 + r2;
      out[b * NSd + o] = x[b * (NSd + NGd + 1) + o] + v * (1.0f / Ld) + pb2[o];
    }
  }
}

extern "C" void kernel_launch(void* const* d_in, const int* in_sizes, int n_in,
                              void* d_out, int out_size, void* d_ws, size_t ws_size,
                              hipStream_t stream) {
  (void)in_sizes; (void)n_in; (void)out_size; (void)ws_size;
  const float* x     = (const float*)d_in[0];
  const float* tss   = (const float*)d_in[1];
  const float* cw1   = (const float*)d_in[2];
  const float* cb1   = (const float*)d_in[3];
  const float* cw2   = (const float*)d_in[4];
  const float* cb2   = (const float*)d_in[5];
  const float* lw    = (const float*)d_in[6];
  const float* lb    = (const float*)d_in[7];
  const float* specw = (const float*)d_in[8];
  const float* convw = (const float*)d_in[9];
  const float* convb = (const float*)d_in[10];
  const float* lng   = (const float*)d_in[11];
  const float* lnb   = (const float*)d_in[12];
  const float* pw1   = (const float*)d_in[13];
  const float* pb1   = (const float*)d_in[14];
  const float* pw2   = (const float*)d_in[15];
  const float* pb2   = (const float*)d_in[16];
  float* out = (float*)d_out;

  char* ws = (char*)d_ws;
  size_t off = 0;
  auto alloc = [&](size_t bytes) -> char* {
    char* p = ws + off;
    off += (bytes + 255) & ~(size_t)255;
    return p;
  };
  bf16_t* fr_il  = (bf16_t*)alloc((size_t)Bd * Wd * Ld * 2);
  bf16_t* fr_li  = (bf16_t*)alloc((size_t)Bd * Ld * Wd * 2);
  bf16_t* FT     = (bf16_t*)alloc((size_t)64 * Bd * Wd * 2);
  bf16_t* OUTft  = (bf16_t*)alloc((size_t)Bd * Wd * 64 * 2);
  bf16_t* dftB   = (bf16_t*)alloc((size_t)64 * Ld * 2);
  bf16_t* idtB   = (bf16_t*)alloc((size_t)Ld * 64 * 2);
  float*  pe     = (float*) alloc((size_t)Wd * Ld * 4);
  float*  lifted = (float*) alloc((size_t)Bd * Wd * 4);
  bf16_t* Wre    = (bf16_t*)alloc((size_t)NBd * Md * Wd * Wd * 2);
  bf16_t* Wim    = (bf16_t*)alloc((size_t)NBd * Md * Wd * Wd * 2);
  bf16_t* Wimn   = (bf16_t*)alloc((size_t)NBd * Md * Wd * Wd * 2);
  bf16_t* cwbf   = (bf16_t*)alloc((size_t)NBd * Wd * Wd * 2);
  bf16_t* p1bf   = (bf16_t*)alloc((size_t)64 * 128 * 2);
  bf16_t* p2bf   = (bf16_t*)alloc((size_t)64 * 64 * 2);

  k_basis<<<4, 256, 0, stream>>>(dftB, idtB, pe);
  long total = (long)NBd * Md * Wd * Wd + (long)NBd * Wd * Wd + 64 * 128 + 64 * 64;
  k_wprep<<<(int)((total + 255) / 256), 256, 0, stream>>>(
      specw, convw, pw1, pw2, Wre, Wim, Wimn, cwbf, p1bf, p2bf);
  k_cond<<<Bd, 128, 0, stream>>>(x, tss, cw1, cb1, cw2, cb2, lw, lb, lifted);
  k_fr0<<<(Bd * Wd * Ld) / 256, 256, 0, stream>>>(lifted, pe, fr_il, fr_li);
  for (int nb = 0; nb < NBd; ++nb) {
    size_t wo = (size_t)nb * Md * Wd * Wd;
    k_dft<<<128, 256, 0, stream>>>(fr_il, dftB, FT);
    k_mix<<<512, 256, 0, stream>>>(FT, Wre + wo, Wim + wo, Wimn + wo, OUTft);
    k_block<<<Bd * 64, 256, 0, stream>>>(OUTft, idtB, cwbf + (size_t)nb * Wd * Wd,
                                         convb + nb * Wd, lng + nb * Wd,
                                         lnb + nb * Wd, fr_il, fr_li);
  }
  k_proj<<<Bd, 128, 0, stream>>>(fr_li, p1bf, pb1, p2bf, pb2, x, out);
}
